// GruRNN_65171833749535
// MI455X (gfx1250) — matrix operand order repaired and checked by reading, not compiled, not run on hardware
//
#include <hip/hip_runtime.h>
#include <hip/hip_bf16.h>

// ---------------------------------------------------------------------------
// GRU forward for MI455X (gfx1250, wave32, WMMA bf16 16x16x32, f32 accum).
//   B=128, T=512, I=H=1024, O=512.
// Pipeline:
//   0) convert xs -> bf16 ; transpose-convert W_ih/W_hh/W_out -> bf16 [N][K]
//   1) x_gates = xs @ W_ih + b_ih          (WMMA GEMM, bf16 out, 384MB ws)
//   2) persistent scan kernel, 512 steps:  hg = h @ W_hh ; fused GRU update
//      (32 workgroups, grid barrier per step; h kept f32 + bf16 shadow)
//   3) out = h_final @ W_out + b_out       (WMMA GEMM, f32 out)
// ---------------------------------------------------------------------------

typedef __bf16 bf16;
typedef __attribute__((ext_vector_type(8)))  __bf16 v8bf;
typedef __attribute__((ext_vector_type(16))) __bf16 v16bf;
typedef __attribute__((ext_vector_type(8)))  float  v8f;
typedef __attribute__((ext_vector_type(4)))  int    v4i;

#define B_ 128
#define T_ 512
#define I_ 1024
#define H_ 1024
#define O_ 512
#define G_ (3 * H_)

// ----- CDNA5 async global->LDS copy (ASYNCcnt) with portable fallback ------
#if defined(__has_builtin)
#if __has_builtin(__builtin_amdgcn_global_load_async_to_lds_b128)
#define HAVE_ASYNC_LDS 1
#endif
#endif

__device__ __forceinline__ void cp16(const void* g, void* l) {
#ifdef HAVE_ASYNC_LDS
  // Probe-confirmed: param 0 is `int __vector(4) *` (generic AS).
  __builtin_amdgcn_global_load_async_to_lds_b128((v4i*)g, (v4i*)l, 0, 0);
#else
  *(v4i*)l = *(const v4i*)g;
#endif
}

__device__ __forceinline__ void wait_async_all() {
#ifdef HAVE_ASYNC_LDS
#if __has_builtin(__builtin_amdgcn_s_wait_asynccnt)
  __builtin_amdgcn_s_wait_asynccnt(0);
#else
  asm volatile("s_wait_asynccnt 0x0" ::: "memory");
#endif
#endif
}

__device__ __forceinline__ void wait_async_4() {
#ifdef HAVE_ASYNC_LDS
#if __has_builtin(__builtin_amdgcn_s_wait_asynccnt)
  __builtin_amdgcn_s_wait_asynccnt(4);
#else
  asm volatile("s_wait_asynccnt 0x4" ::: "memory");
#endif
#endif
}

// A-fragment (16x32 bf16): lane<16 -> K[0..15];
// lane>=16 -> K[16..31]. base points at row m, kbase = 0 or 16.
__device__ __forceinline__ v16bf load_afrag(const bf16* base, int kbase) {
  v8bf lo = *(const v8bf*)(base + kbase);
  v8bf hi = *(const v8bf*)(base + kbase + 8);
  return __builtin_shufflevector(lo, hi, 0, 1, 2, 3, 4, 5, 6, 7,
                                 8, 9, 10, 11, 12, 13, 14, 15);
}

// ---------------------------------------------------------------------------
// Elementwise conversion / transpose / init kernels
// ---------------------------------------------------------------------------
__global__ __launch_bounds__(256) void cvt_bf16_kernel(
    const float* __restrict__ in, bf16* __restrict__ out, size_t n) {
  size_t i = (size_t)blockIdx.x * blockDim.x + threadIdx.x;
  size_t stride = (size_t)gridDim.x * blockDim.x;
  for (; i < n; i += stride) out[i] = (bf16)in[i];
}

// W: [K][N] f32 row-major -> WT: [N][K] bf16 (K contiguous)
__global__ __launch_bounds__(256) void transpose_cvt_kernel(
    const float* __restrict__ W, bf16* __restrict__ WT, int K, int N) {
  size_t total = (size_t)K * N;
  size_t i = (size_t)blockIdx.x * blockDim.x + threadIdx.x;
  size_t stride = (size_t)gridDim.x * blockDim.x;
  for (; i < total; i += stride) {
    size_t k = i / (size_t)N, n = i % (size_t)N;
    WT[n * (size_t)K + k] = (bf16)W[i];
  }
}

__global__ __launch_bounds__(256) void scan_init_kernel(
    bf16* __restrict__ h, float* __restrict__ hf, unsigned* __restrict__ bar) {
  size_t i = (size_t)blockIdx.x * blockDim.x + threadIdx.x;
  if (i < (size_t)B_ * H_) {
    hf[i] = 0.f;
    h[i] = (bf16)0.f;
    h[(size_t)B_ * H_ + i] = (bf16)0.f;
  }
  if (i == 0) bar[0] = 0u;
}

// ---------------------------------------------------------------------------
// Generic WMMA GEMM:  C[M x N] = A[M x K] * BT[N x K]^T + bias
// A, BT bf16 (K contiguous). 128x128 tile / workgroup, 8 waves, wave = 32x64.
// Double-buffered LDS, async staging overlapped with WMMA compute.
// ---------------------------------------------------------------------------
template <bool OUTF32>
__global__ __launch_bounds__(256) void wmma_gemm(
    const bf16* __restrict__ A, const bf16* __restrict__ BT,
    const float* __restrict__ bias, void* __restrict__ Cv, int K, int ldc) {
  __shared__ bf16 sA[2][128 * 32];
  __shared__ bf16 sB[2][128 * 32];

  const int tid = threadIdx.x;
  const int lane = tid & 31;
  const int w = tid >> 5;
  const int wm = w >> 1;              // 0..3 (M)
  const int wn = w & 1;               // 0..1 (N)
  const int l16 = lane & 15;
  const int hi8 = (lane >> 4) * 8;    // C row offset (lane>=16 -> M+8)
  const int hi16 = (lane >> 4) * 16;  // B K-phase
  const long tileM = (long)blockIdx.y * 128;
  const long tileN = (long)blockIdx.x * 128;

  v8f acc[2][4];
#pragma unroll
  for (int i = 0; i < 2; ++i)
#pragma unroll
    for (int j = 0; j < 4; ++j) acc[i][j] = {};

  auto stage = [&](int buf, int kk) {
#pragma unroll
    for (int i = 0; i < 2; ++i) {  // A: 512 16B chunks / 256 threads
      int ca = tid * 2 + i;
      int row = ca >> 2, kc = (ca & 3) * 8;
      cp16(A + (tileM + row) * (long)K + (long)kk * 32 + kc,
           &sA[buf][row * 32 + kc]);
    }
#pragma unroll
    for (int i = 0; i < 2; ++i) {  // B: 512 16B chunks / 256 threads
      int cb = tid * 2 + i;
      int row = cb >> 2, kc = (cb & 3) * 8;
      cp16(BT + (tileN + row) * (long)K + (long)kk * 32 + kc,
           &sB[buf][row * 32 + kc]);
    }
  };

  const int KT = K >> 5;
  stage(0, 0);
  for (int kk = 0; kk < KT; ++kk) {
    const int cur = kk & 1;
    if (kk + 1 < KT) {
      stage(cur ^ 1, kk + 1);  // issue next tile async
      wait_async_4();          // wait only for the current tile's 4 chunks
    } else {
      wait_async_all();
    }
    __syncthreads();

    const bf16* At = sA[cur];
    const bf16* Bt = sB[cur];
    v16bf a[2], b[4];
#pragma unroll
    for (int ms = 0; ms < 2; ++ms)
      a[ms] = load_afrag(At + (wm * 32 + ms * 16 + l16) * 32, hi16);
#pragma unroll
    for (int ns = 0; ns < 4; ++ns)
      b[ns] = *(const v16bf*)(Bt + (wn * 64 + ns * 16 + l16) * 32 + hi16);
#pragma unroll
    for (int ms = 0; ms < 2; ++ms)
#pragma unroll
      for (int ns = 0; ns < 4; ++ns)
        acc[ms][ns] = __builtin_amdgcn_wmma_f32_16x16x32_bf16(
            false, a[ms], false, b[ns], (short)0, acc[ms][ns], false, false);
    __syncthreads();
  }

  float* Cf = (float*)Cv;
  bf16* Cb = (bf16*)Cv;
#pragma unroll
  for (int ms = 0; ms < 2; ++ms) {
    long mbase = tileM + wm * 32 + ms * 16 + hi8;  // C: lane>=16 -> M+8
#pragma unroll
    for (int ns = 0; ns < 4; ++ns) {
      long n = tileN + wn * 64 + ns * 16 + l16;
      float bv = bias[n];
      v8f c = acc[ms][ns];
#pragma unroll
      for (int r = 0; r < 8; ++r) {
        float v = c[r] + bv;
        long idx = (mbase + r) * (long)ldc + n;
        if (OUTF32) Cf[idx] = v; else Cb[idx] = (bf16)v;
      }
    }
  }
}

// ---------------------------------------------------------------------------
// Persistent GRU scan. 32 workgroups x 192 threads (6 waves).
// WG owns 32 h-columns; wave = (gate, M-half): 64x32 = 4x2 WMMA accumulators.
// Per step: hg = h_bf16 @ W_hh^T (K=1024), gates exchanged via LDS, fused
// sigmoid/tanh update, h written f32 + bf16, then device-scope grid barrier.
// ---------------------------------------------------------------------------
__global__ __launch_bounds__(192) void gru_scan(
    const bf16* __restrict__ xg, const bf16* __restrict__ WhhT,
    const float* __restrict__ bhh, bf16* __restrict__ hbuf,
    float* __restrict__ hf, unsigned* __restrict__ bar) {
  __shared__ bf16 sH[128 * 32];        // h tile  [m][k]   8 KB
  __shared__ bf16 sW[96 * 32];         // W^T tile [j][k]  6 KB
  __shared__ float sHG[3 * 128 * 32];  // gate results     48 KB

  const int tid = threadIdx.x;
  const int lane = tid & 31;
  const int w = tid >> 5;
  const int g = w >> 1;   // gate 0..2 (r,z,n)
  const int mh = w & 1;   // M half
  const int l16 = lane & 15;
  const int hi8 = (lane >> 4) * 8;
  const int hi16 = (lane >> 4) * 16;
  const int nc0 = blockIdx.x * 32;  // owned h-column range
  const unsigned nwg = gridDim.x;

  for (int t = 0; t < T_; ++t) {
    const bf16* hcur = hbuf + (size_t)(t & 1) * (B_ * H_);
    bf16* hnext = hbuf + (size_t)((t + 1) & 1) * (B_ * H_);

    v8f acc[4][2];
#pragma unroll
    for (int i = 0; i < 4; ++i)
#pragma unroll
      for (int j = 0; j < 2; ++j) acc[i][j] = {};

    for (int kk = 0; kk < H_ / 32; ++kk) {
      // --- stage: waves 0..3 copy h tile, waves 4..5 copy W_hh^T tile ---
      if (tid < 128) {
        const int m = tid;
#pragma unroll
        for (int c = 0; c < 4; ++c)
          cp16(hcur + (size_t)m * H_ + kk * 32 + c * 8, &sH[m * 32 + c * 8]);
      } else {
        const int q = tid - 128;
#pragma unroll
        for (int i = 0; i < 6; ++i) {
          int cc = q * 6 + i;  // 0..383
          int j = cc >> 2, c = cc & 3;
          int gg = j >> 5, jj = j & 31;
          cp16(WhhT + (size_t)(gg * H_ + nc0 + jj) * H_ + kk * 32 + c * 8,
               &sW[j * 32 + c * 8]);
        }
      }
      wait_async_all();
      __syncthreads();

      v16bf a[4], b[2];
#pragma unroll
      for (int ms = 0; ms < 4; ++ms)
        a[ms] = load_afrag(&sH[(mh * 64 + ms * 16 + l16) * 32], hi16);
#pragma unroll
      for (int ns = 0; ns < 2; ++ns)
        b[ns] = *(const v16bf*)&sW[(g * 32 + ns * 16 + l16) * 32 + hi16];
#pragma unroll
      for (int ms = 0; ms < 4; ++ms)
#pragma unroll
        for (int ns = 0; ns < 2; ++ns)
          acc[ms][ns] = __builtin_amdgcn_wmma_f32_16x16x32_bf16(
              false, a[ms], false, b[ns], (short)0, acc[ms][ns], false, false);
      __syncthreads();
    }

    // --- publish per-gate results to LDS for the fused update ---
#pragma unroll
    for (int ms = 0; ms < 4; ++ms) {
      int mb = mh * 64 + ms * 16 + hi8;
#pragma unroll
      for (int ns = 0; ns < 2; ++ns) {
        int j = ns * 16 + l16;
#pragma unroll
        for (int r = 0; r < 8; ++r)
          sHG[(g * 128 + mb + r) * 32 + j] = acc[ms][ns][r];
      }
    }
    __syncthreads();

    // --- fused GRU update for the 128x32 owned block ---
    for (int e = tid; e < 128 * 32; e += 192) {
      int m = e >> 5, j = e & 31;
      int col = nc0 + j;
      float hr = sHG[(0 * 128 + m) * 32 + j] + bhh[col];
      float hz = sHG[(1 * 128 + m) * 32 + j] + bhh[H_ + col];
      float hn = sHG[(2 * 128 + m) * 32 + j] + bhh[2 * H_ + col];
      size_t xb = ((size_t)m * T_ + t) * G_;
      float xr = (float)xg[xb + col];
      float xz = (float)xg[xb + H_ + col];
      float xn = (float)xg[xb + 2 * H_ + col];
      float rr = 1.f / (1.f + __expf(-(xr + hr)));
      float zz = 1.f / (1.f + __expf(-(xz + hz)));
      float nn = tanhf(xn + rr * hn);
      float hold = hf[(size_t)m * H_ + col];
      float hnew = (1.f - zz) * nn + zz * hold;
      hf[(size_t)m * H_ + col] = hnew;
      hnext[(size_t)m * H_ + col] = (bf16)hnew;
    }

    // --- device-scope grid barrier (monotonic counter, s_sleep poll) ---
    __threadfence();
    __syncthreads();
    if (tid == 0) {
      __hip_atomic_fetch_add(bar, 1u, __ATOMIC_RELEASE,
                             __HIP_MEMORY_SCOPE_AGENT);
      const unsigned target = (unsigned)(t + 1) * nwg;
      while (__hip_atomic_load(bar, __ATOMIC_ACQUIRE,
                               __HIP_MEMORY_SCOPE_AGENT) < target)
        __builtin_amdgcn_s_sleep(8);
    }
    __syncthreads();
  }
}

// ---------------------------------------------------------------------------
extern "C" void kernel_launch(void* const* d_in, const int* in_sizes, int n_in,
                              void* d_out, int out_size, void* d_ws,
                              size_t ws_size, hipStream_t stream) {
  (void)in_sizes; (void)n_in; (void)out_size; (void)ws_size;
  const float* xs = (const float*)d_in[0];
  const float* Wih = (const float*)d_in[1];
  const float* bih = (const float*)d_in[2];
  const float* Whh = (const float*)d_in[3];
  const float* bhh = (const float*)d_in[4];
  const float* Wout = (const float*)d_in[5];
  const float* bout = (const float*)d_in[6];

  char* ws = (char*)d_ws;
  size_t off = 0;
  auto alloc = [&](size_t bytes) {
    void* p = ws + off;
    off = (off + bytes + 255) & ~(size_t)255;
    return p;
  };
  bf16* xs_bf = (bf16*)alloc((size_t)B_ * T_ * I_ * 2);   // 128 MB
  bf16* WihT  = (bf16*)alloc((size_t)I_ * G_ * 2);        //   6 MB
  bf16* WhhT  = (bf16*)alloc((size_t)H_ * G_ * 2);        //   6 MB
  bf16* WoutT = (bf16*)alloc((size_t)H_ * O_ * 2);        //   1 MB
  bf16* xgate = (bf16*)alloc((size_t)B_ * T_ * G_ * 2);   // 384 MB
  bf16* hbuf  = (bf16*)alloc((size_t)2 * B_ * H_ * 2);    // 512 KB
  float* hf32 = (float*)alloc((size_t)B_ * H_ * 4);       // 512 KB
  unsigned* bar = (unsigned*)alloc(256);

  // 0) precision conversion / weight transposes
  cvt_bf16_kernel<<<8192, 256, 0, stream>>>(xs, xs_bf, (size_t)B_ * T_ * I_);
  transpose_cvt_kernel<<<4096, 256, 0, stream>>>(Wih, WihT, I_, G_);
  transpose_cvt_kernel<<<4096, 256, 0, stream>>>(Whh, WhhT, H_, G_);
  transpose_cvt_kernel<<<1024, 256, 0, stream>>>(Wout, WoutT, H_, O_);
  scan_init_kernel<<<(B_ * H_ + 255) / 256, 256, 0, stream>>>(hbuf, hf32, bar);

  // 1) x_gates = xs @ W_ih + b_ih   [65536 x 3072], bf16 output
  wmma_gemm<false><<<dim3(G_ / 128, (B_ * T_) / 128), 256, 0, stream>>>(
      xs_bf, WihT, bih, (void*)xgate, I_, G_);

  // 2) 512-step recurrent scan (persistent kernel, grid barrier per step)
  gru_scan<<<H_ / 32, 192, 0, stream>>>(xgate, WhhT, bhh, hbuf, hf32, bar);

  // 3) out = h_final @ W_out + b_out  (final h lands in hbuf[0]; T even)
  wmma_gemm<true><<<dim3(O_ / 128, B_ / 128), 256, 0, stream>>>(
      hbuf, WoutT, bout, d_out, H_, O_);
}
